// GCN_LSTM_81784767251211
// MI455X (gfx1250) — matrix operand order repaired and checked by reading, not compiled, not run on hardware
//
#include <hip/hip_runtime.h>
#include <hip/hip_bf16.h>
#include <cmath>

// ---------------------------------------------------------------------------
// Problem constants (match reference)
// ---------------------------------------------------------------------------
#define Bc   16
#define Tc   12
#define Nc   4000
#define Ec   64000
#define GHc  32
#define LHc  64
#define BTc  (Bc * Tc)          // 192
#define ROWS (Bc * Nc)          // 64000 LSTM rows
#define G4   (4 * LHc)          // 256 gate width

typedef _Float16 half_t;
typedef half_t v8h  __attribute__((ext_vector_type(8)));
typedef half_t v16h __attribute__((ext_vector_type(16)));
typedef float  v8f  __attribute__((ext_vector_type(8)));

__device__ __forceinline__ v8f wmma_f16(v16h a, v16h b, v8f c) {
    // D = A(16x32 f16) * B(32x16 f16) + C(16x16 f32)
    return __builtin_amdgcn_wmma_f32_16x16x32_f16(
        /*neg_a=*/false, a, /*neg_b=*/false, b,
        /*c_mod=*/(short)0, c, /*reuse_a=*/false, /*reuse_b=*/false);
}

// Async copy of one contiguous 1 KB tile (32 lanes x 16 B x 2 instructions).
// IOFFSET applies to both the LDS and global addresses (ISA 15.18.3), so the
// second instruction reuses the same address registers with offset:512.
__device__ __forceinline__ void async_copy_1k(void* lds, const void* gmem, int lane) {
    unsigned l = (unsigned)(size_t)lds + (unsigned)lane * 16u;   // LDS byte addr
    const char* g = (const char*)gmem + lane * 16;
    asm volatile(
        "global_load_async_to_lds_b128 %0, %1, off\n\t"
        "global_load_async_to_lds_b128 %0, %1, off offset:512"
        :
        : "v"(l), "v"(g)
        : "memory");
}

__device__ __forceinline__ void async_drain() {
    asm volatile("s_wait_asynccnt 0x0" ::: "memory");
}

// ---------------------------------------------------------------------------
// Graph preprocessing: degrees (with self loops), dinv, per-edge norm, CSR
// ---------------------------------------------------------------------------
__global__ void init_kernel(float* deg, int* cnt) {
    int n = blockIdx.x * 256 + threadIdx.x;
    if (n < Nc) { deg[n] = 1.0f; cnt[n] = 0; }   // self-loop weight 1
}

__global__ void edge_acc_kernel(const int* __restrict__ ei,
                                const float* __restrict__ w,
                                float* deg, int* cnt) {
    int e = blockIdx.x * 256 + threadIdx.x;
    if (e < Ec) {
        int d = ei[Ec + e];
        atomicAdd(&deg[d], w[e]);
        atomicAdd(&cnt[d], 1);
    }
}

__global__ void dinv_kernel(const float* __restrict__ deg, float* dinv) {
    int n = blockIdx.x * 256 + threadIdx.x;
    if (n < Nc) { float d = deg[n]; dinv[n] = d > 0.f ? rsqrtf(d) : 0.f; }
}

__global__ void norm_kernel(const int* __restrict__ ei,
                            const float* __restrict__ ew,
                            const float* __restrict__ dinv,
                            float* nrm) {
    int e = blockIdx.x * 256 + threadIdx.x;
    if (e < Ec) nrm[e] = dinv[ei[e]] * ew[e] * dinv[ei[Ec + e]];
}

// exclusive scan of 4000 counts in a single 256-thread block
__global__ void scan_kernel(const int* __restrict__ cnt, int* offs, int* cur) {
    __shared__ int tsum[256];
    int tid = threadIdx.x;
    int base = tid * 16;
    int loc[16]; int s = 0;
    #pragma unroll
    for (int i = 0; i < 16; i++) {
        int c = (base + i < Nc) ? cnt[base + i] : 0;
        loc[i] = s; s += c;
    }
    tsum[tid] = s;
    __syncthreads();
    if (tid == 0) {
        int a = 0;
        for (int j = 0; j < 256; j++) { int t = tsum[j]; tsum[j] = a; a += t; }
    }
    __syncthreads();
    int pfx = tsum[tid];
    #pragma unroll
    for (int i = 0; i < 16; i++) {
        int n = base + i;
        if (n < Nc) { int o = pfx + loc[i]; offs[n] = o; cur[n] = o; }
    }
    if (tid == 255) offs[Nc] = Ec;
}

__global__ void fill_kernel(const int* __restrict__ ei,
                            const float* __restrict__ nrm,
                            int* cur, int* csrc, float* cnrm) {
    int e = blockIdx.x * 256 + threadIdx.x;
    if (e < Ec) {
        int s = ei[e], d = ei[Ec + e];
        int pos = atomicAdd(&cur[d], 1);
        csrc[pos] = s;
        cnrm[pos] = nrm[e];
    }
}

// ---------------------------------------------------------------------------
// Weight prep: f16 out-major weights (WMMA B layout reads contiguous K runs)
// ---------------------------------------------------------------------------
__global__ void prep_kernel(const float* __restrict__ Wih,   // [256][32]
                            const float* __restrict__ Whh,   // [256][64]
                            const float* __restrict__ W2,    // [32][32] in-major
                            const float* __restrict__ bih,
                            const float* __restrict__ bhh,
                            half_t* wih16, half_t* whh16, half_t* w2t16,
                            float* biasc) {
    int tid = threadIdx.x;
    for (int i = tid; i < G4 * GHc; i += 256) wih16[i] = (half_t)Wih[i];
    for (int i = tid; i < G4 * LHc; i += 256) whh16[i] = (half_t)Whh[i];
    for (int i = tid; i < GHc * GHc; i += 256) {
        int o = i >> 5, k = i & 31;
        w2t16[o * GHc + k] = (half_t)W2[k * GHc + o];   // transpose -> out-major
    }
    if (tid < G4) biasc[tid] = bih[tid] + bhh[tid];
}

// ---------------------------------------------------------------------------
// Layer-1 scalar aggregation: s1[n][bt] = sum_e norm*x[bt,src] + dinv^2*x[bt,n]
// (CIN=1 => aggregation commutes with the rank-1 linear map)
// ---------------------------------------------------------------------------
__global__ void s1_kernel(const float* __restrict__ x,
                          const int* __restrict__ offs,
                          const int* __restrict__ csrc,
                          const float* __restrict__ cnrm,
                          const float* __restrict__ dinv,
                          float* __restrict__ s1) {
    int n = blockIdx.x;          // 0..3999
    int bt = threadIdx.x;        // 0..191
    int beg = offs[n], end = offs[n + 1];
    const float* xb = x + (size_t)bt * Nc;
    float s = 0.f;
    for (int i = beg; i < end; i++) s += cnrm[i] * xb[csrc[i]];
    float dn = dinv[n];
    s += dn * dn * xb[n];
    s1[n * BTc + bt] = s;
}

// ---------------------------------------------------------------------------
// lin2 = relu(s1*W1 + b1) @ W2 via WMMA.  1 wave per 16-row tile, A built in
// registers, 2 N-tiles of W2 -> 2 WMMAs per tile.
// ---------------------------------------------------------------------------
__global__ __launch_bounds__(128) void lin2_kernel(
        const float* __restrict__ s1,
        const float* __restrict__ W1, const float* __restrict__ b1,
        const half_t* __restrict__ W2T,
        float* __restrict__ lin2) {
    __shared__ float w1s[GHc], b1s[GHc];
    int tid = threadIdx.x;
    if (tid < GHc) { w1s[tid] = W1[tid]; b1s[tid] = b1[tid]; }
    __syncthreads();

    int tile = blockIdx.x * 4 + (tid >> 5);   // 48000 tiles total
    int lane = tid & 31;
    int bt = tile / 250;
    int n0 = (tile % 250) * 16;
    int rr = lane & 15;
    int hi = lane >> 4;                       // 0: K 0..15  1: K 16..31

    float s = s1[(n0 + rr) * BTc + bt];
    v16h a;
    int kbase = hi * 16;
    #pragma unroll
    for (int i = 0; i < 16; i++) {
        int k = kbase + i;  // {kb..kb+15}
        float v = s * w1s[k] + b1s[k];
        a[i] = (half_t)(v > 0.f ? v : 0.f);
    }

    #pragma unroll
    for (int nt = 0; nt < 2; nt++) {
        v16h bfr = *(const v16h*)(W2T + (nt * 16 + rr) * GHc + hi * 16);
        v8f acc = {};
        acc = wmma_f16(a, bfr, acc);
        #pragma unroll
        for (int v = 0; v < 8; v++) {
            int m = v + 8 * hi;
            lin2[((size_t)(bt * Nc + n0 + m)) * GHc + nt * 16 + rr] = acc[v];
        }
    }
}

// ---------------------------------------------------------------------------
// Layer-2 aggregation (CSR gather, lane = channel) + self loop + bias + relu.
// Output stored directly in f16 (WMMA A-operand precision): halves the
// GCN->LSTM intermediate to 49 MB and makes LSTM staging a pure byte copy.
// ---------------------------------------------------------------------------
__global__ __launch_bounds__(128) void agg2_kernel(
        const float* __restrict__ lin2,
        const int* __restrict__ offs,
        const int* __restrict__ csrc,
        const float* __restrict__ cnrm,
        const float* __restrict__ dinv,
        const float* __restrict__ b2,
        half_t* __restrict__ h2) {
    int id = blockIdx.x * 4 + (threadIdx.x >> 5);   // 0..767999 = (bt, n)
    int lane = threadIdx.x & 31;
    int bt = id / Nc;
    int n  = id - bt * Nc;
    size_t rowbase = (size_t)bt * Nc;
    int beg = offs[n], end = offs[n + 1];
    float acc = 0.f;
    for (int i = beg; i < end; i++) {
        int src = csrc[i];
        float w = cnrm[i];
        acc += w * lin2[(rowbase + src) * GHc + lane];
    }
    float dn = dinv[n];
    acc += dn * dn * lin2[(rowbase + n) * GHc + lane];
    acc += b2[lane];
    h2[(rowbase + n) * GHc + lane] = (half_t)(acc > 0.f ? acc : 0.f);
}

// ---------------------------------------------------------------------------
// Fused LSTM + FC. One block = 16 rows (nodes), 4 waves.
// Wave w owns 4 gate N-tiles: 3 WMMAs each (X*Wih^T + two K-halves of H*Whh^T).
// Weight B-fragments persist in VGPRs across all 12 timesteps.
// x_t tiles (1 KB contiguous f16) are double-buffered in LDS via
// GLOBAL_LOAD_ASYNC_TO_LDS_B128 issued by wave 0 one step ahead.
// ---------------------------------------------------------------------------
__global__ __launch_bounds__(128) void lstm_kernel(
        const half_t* __restrict__ hseq,  // [B*T][N][32] f16
        const half_t* __restrict__ Wih,   // [256][32] f16
        const half_t* __restrict__ Whh,   // [256][64] f16
        const float* __restrict__ biasc,  // [256]
        const float* __restrict__ Wfc,    // [64]
        const float* __restrict__ bfc,    // [1]
        float* __restrict__ out) {        // [B*N]
    __shared__ __align__(64) half_t xbuf[2][16][GHc];   // double-buffered x_t
    __shared__ __align__(32) half_t hlds[16][LHc];
    __shared__ __align__(16) float  gl[16][G4];
    __shared__ float hfin[16][LHc];
    __shared__ float bs[G4];
    __shared__ float wfcs[LHc];

    int tid  = threadIdx.x;
    int wv   = tid >> 5;
    int lane = tid & 31;
    int rr   = lane & 15;
    int hi   = lane >> 4;

    int tileg = blockIdx.x;               // 0..3999 (16 rows each)
    int b  = (tileg * 16) / Nc;
    int n0 = (tileg * 16) % Nc;

    if (tid < LHc) wfcs[tid] = Wfc[tid];
    for (int i = tid; i < G4; i += 128) bs[i] = biasc[i];
    for (int i = tid; i < 16 * LHc; i += 128) ((half_t*)hlds)[i] = (half_t)0.f;

    float cst[8];
    #pragma unroll
    for (int k = 0; k < 8; k++) cst[k] = 0.f;
    int er = tid >> 3;            // elementwise row
    int ej = (tid & 7) * 8;       // elementwise col base

    // Persistent B fragments (weights): 4 N-tiles * (1 ih + 2 hh) per wave
    v16h bih[4], bhh[4][2];
    #pragma unroll
    for (int q = 0; q < 4; q++) {
        int row = (wv * 4 + q) * 16 + rr;
        bih[q]    = *(const v16h*)(Wih + row * GHc + hi * 16);
        bhh[q][0] = *(const v16h*)(Whh + row * LHc + hi * 16);
        bhh[q][1] = *(const v16h*)(Whh + row * LHc + 32 + hi * 16);
    }

    // tile t in global memory: contiguous 1 KB at rows [n0, n0+16)
    const half_t* gtile0 = hseq + ((size_t)(b * Tc + 0) * Nc + n0) * GHc;

    // prefetch t=0 into xbuf[0]
    if (wv == 0) {
        async_copy_1k(&xbuf[0][0][0], gtile0, lane);
        async_drain();
    }
    __syncthreads();

    for (int t = 0; t < Tc; t++) {
        int cur = t & 1;
        // kick off async staging of x_{t+1} while computing step t
        if (wv == 0 && t + 1 < Tc) {
            const half_t* gt = hseq + ((size_t)(b * Tc + (t + 1)) * Nc + n0) * GHc;
            async_copy_1k(&xbuf[cur ^ 1][0][0], gt, lane);
        }

        // A fragments from LDS
        v8h x0 = *(const v8h*)(&xbuf[cur][rr][hi * 16]);
        v8h x1 = *(const v8h*)(&xbuf[cur][rr][hi * 16 + 8]);
        v16h ax = __builtin_shufflevector(x0, x1, 0,1,2,3,4,5,6,7,8,9,10,11,12,13,14,15);
        v8h h0 = *(const v8h*)(&hlds[rr][hi * 16]);
        v8h h1 = *(const v8h*)(&hlds[rr][hi * 16 + 8]);
        v16h ah0 = __builtin_shufflevector(h0, h1, 0,1,2,3,4,5,6,7,8,9,10,11,12,13,14,15);
        v8h h2a = *(const v8h*)(&hlds[rr][32 + hi * 16]);
        v8h h2b = *(const v8h*)(&hlds[rr][32 + hi * 16 + 8]);
        v16h ah1 = __builtin_shufflevector(h2a, h2b, 0,1,2,3,4,5,6,7,8,9,10,11,12,13,14,15);

        #pragma unroll
        for (int q = 0; q < 4; q++) {
            v8f acc = {};
            acc = wmma_f16(ax,  bih[q],    acc);
            acc = wmma_f16(ah0, bhh[q][0], acc);
            acc = wmma_f16(ah1, bhh[q][1], acc);
            int nt = wv * 4 + q;
            #pragma unroll
            for (int v = 0; v < 8; v++)
                gl[v + 8 * hi][nt * 16 + rr] = acc[v];
        }
        __syncthreads();

        // gate nonlinearity + state update (torch gate order i,f,g,o)
        #pragma unroll
        for (int k = 0; k < 8; k++) {
            int j = ej + k;
            float ig = gl[er][j]           + bs[j];
            float fg = gl[er][LHc + j]     + bs[LHc + j];
            float gg = gl[er][2 * LHc + j] + bs[2 * LHc + j];
            float og = gl[er][3 * LHc + j] + bs[3 * LHc + j];
            float is = 1.f / (1.f + __expf(-ig));
            float fs = 1.f / (1.f + __expf(-fg));
            float os = 1.f / (1.f + __expf(-og));
            cst[k] = fs * cst[k] + is * tanhf(gg);
            float hv = os * tanhf(cst[k]);
            hlds[er][j] = (half_t)hv;
            if (t == Tc - 1) hfin[er][j] = hv;
        }
        // drain the async stage for x_{t+1} before everyone passes the barrier
        if (wv == 0 && t + 1 < Tc) async_drain();
        __syncthreads();
    }

    // final FC: out[row] = h_T . Wfc + b_fc
    if (tid < 16) {
        float s = 0.f;
        for (int j = 0; j < LHc; j++) s += hfin[tid][j] * wfcs[j];
        out[b * Nc + n0 + tid] = s + bfc[0];
    }
}

// ---------------------------------------------------------------------------
// Launch
// ---------------------------------------------------------------------------
extern "C" void kernel_launch(void* const* d_in, const int* in_sizes, int n_in,
                              void* d_out, int out_size, void* d_ws, size_t ws_size,
                              hipStream_t stream) {
    const float* x    = (const float*)d_in[0];
    const int*   ei   = (const int*)  d_in[1];
    const float* ew   = (const float*)d_in[2];
    const float* W1   = (const float*)d_in[3];
    const float* b1   = (const float*)d_in[4];
    const float* W2   = (const float*)d_in[5];
    const float* b2   = (const float*)d_in[6];
    const float* Wih  = (const float*)d_in[7];
    const float* Whh  = (const float*)d_in[8];
    const float* bih  = (const float*)d_in[9];
    const float* bhh  = (const float*)d_in[10];
    const float* Wfc  = (const float*)d_in[11];
    const float* bfc  = (const float*)d_in[12];
    float* out = (float*)d_out;

    char* p = (char*)d_ws;
    auto alloc = [&](size_t bytes) -> char* {
        char* r = p;
        p += (bytes + 255) & ~(size_t)255;
        return r;
    };
    float*  deg   = (float*)alloc(Nc * 4);
    float*  dinv  = (float*)alloc(Nc * 4);
    float*  nrm   = (float*)alloc(Ec * 4);
    int*    cnt   = (int*)  alloc(Nc * 4);
    int*    offs  = (int*)  alloc((Nc + 1) * 4);
    int*    cur   = (int*)  alloc(Nc * 4);
    int*    csrc  = (int*)  alloc(Ec * 4);
    float*  cnrm  = (float*)alloc(Ec * 4);
    float*  s1    = (float*)alloc((size_t)BTc * Nc * 4);                 // 3 MB
    float*  lin2  = (float*)alloc((size_t)BTc * Nc * GHc * 4);           // 98 MB (L2-resident)
    half_t* h2    = (half_t*)alloc((size_t)BTc * Nc * GHc * 2);          // 49 MB f16 (L2-resident)
    half_t* wih16 = (half_t*)alloc((size_t)G4 * GHc * 2);
    half_t* whh16 = (half_t*)alloc((size_t)G4 * LHc * 2);
    half_t* w2t16 = (half_t*)alloc((size_t)GHc * GHc * 2);
    float*  biasc = (float*)alloc(G4 * 4);

    // graph preprocessing + weight prep
    init_kernel<<<(Nc + 255) / 256, 256, 0, stream>>>(deg, cnt);
    prep_kernel<<<1, 256, 0, stream>>>(Wih, Whh, W2, bih, bhh,
                                       wih16, whh16, w2t16, biasc);
    edge_acc_kernel<<<(Ec + 255) / 256, 256, 0, stream>>>(ei, ew, deg, cnt);
    dinv_kernel<<<(Nc + 255) / 256, 256, 0, stream>>>(deg, dinv);
    norm_kernel<<<(Ec + 255) / 256, 256, 0, stream>>>(ei, ew, dinv, nrm);
    scan_kernel<<<1, 256, 0, stream>>>(cnt, offs, cur);
    fill_kernel<<<(Ec + 255) / 256, 256, 0, stream>>>(ei, nrm, cur, csrc, cnrm);

    // GCN layer 1 (scalar aggregate) + layer 2 linear (WMMA) + layer 2 gather
    s1_kernel<<<Nc, BTc, 0, stream>>>(x, offs, csrc, cnrm, dinv, s1);
    lin2_kernel<<<(BTc * (Nc / 16)) / 4, 128, 0, stream>>>(s1, W1, b1, w2t16, lin2);
    agg2_kernel<<<(BTc * Nc) / 4, 128, 0, stream>>>(lin2, offs, csrc, cnrm, dinv, b2, h2);

    // fused LSTM (WMMA + async double-buffered staging) + FC
    lstm_kernel<<<ROWS / 16, 128, 0, stream>>>(h2, wih16, whh16, biasc, Wfc, bfc, out);
    (void)in_sizes; (void)n_in; (void)out_size; (void)ws_size;
}
